// mLSTM_63496796504585
// MI455X (gfx1250) — matrix operand order repaired and checked by reading, not compiled, not run on hardware
//
#include <hip/hip_runtime.h>

typedef __attribute__((ext_vector_type(16))) _Float16 v16h;
typedef __attribute__((ext_vector_type(8)))  _Float16 v8h;
typedef __attribute__((ext_vector_type(8)))  float    v8f;
typedef __attribute__((ext_vector_type(4))) unsigned int u32x4;
typedef __attribute__((ext_vector_type(8))) int i32x8_t;
typedef __attribute__((ext_vector_type(4))) int i32x4_t;

constexpr int B_   = 256;
constexpr int D_   = 1024;
constexpr int H_   = 8;
constexpr int HD_  = 128;
constexpr int P_   = 2048;
constexpr int O_   = 1024;
constexpr int HID_ = H_ * HD_;
constexpr float EPS_ = 1e-5f;

// GEMM tiling: 128(M) x 64(N) block tile, K staged through LDS in KC slices.
constexpr int KC  = 64;  // K elements per stage
constexpr int LDR = 72;  // padded LDS row stride in halves (64 + 8 pad)

#ifndef USE_TDM
#if defined(__has_builtin)
#if __has_builtin(__builtin_amdgcn_tensor_load_to_lds)
#define USE_TDM 1
#else
#define USE_TDM 0
#endif
#else
#define USE_TDM 0
#endif
#endif

// ---------------------------------------------------------------- reductions
__device__ __forceinline__ float block_reduce_sum(float v, float* red, int tid) {
  red[tid] = v;
  __syncthreads();
  for (int s = 128; s > 0; s >>= 1) {
    if (tid < s) red[tid] += red[tid + s];
    __syncthreads();
  }
  float r = red[0];
  __syncthreads();
  return r;
}

// ---------------------------------------------------------------- LayerNorm
__global__ __launch_bounds__(256) void ln_kernel(
    const float* __restrict__ seq, const float* __restrict__ g,
    const float* __restrict__ b, float* __restrict__ xn,
    _Float16* __restrict__ xnh) {
  __shared__ float red[256];
  const int row = blockIdx.x, tid = threadIdx.x;
  const float* x = seq + (size_t)row * D_;
  float v[4];
  float s = 0.f, ss = 0.f;
#pragma unroll
  for (int i = 0; i < 4; ++i) {
    v[i] = x[tid + i * 256];
    s += v[i];
    ss += v[i] * v[i];
  }
  float sum = block_reduce_sum(s, red, tid);
  float sq  = block_reduce_sum(ss, red, tid);
  float mu  = sum * (1.f / D_);
  float var = sq * (1.f / D_) - mu * mu;
  float inv = rsqrtf(var + EPS_);
#pragma unroll
  for (int i = 0; i < 4; ++i) {
    int j = tid + i * 256;
    float y = (v[i] - mu) * inv * g[j] + b[j];
    xn[(size_t)row * D_ + j]  = y;
    xnh[(size_t)row * D_ + j] = (_Float16)y;
  }
}

// ---------------------------------------------------------------- f32 -> f16
__global__ void cvt_kernel(const float* __restrict__ s, _Float16* __restrict__ d,
                           int n) {
  for (int i = blockIdx.x * blockDim.x + threadIdx.x; i < n;
       i += gridDim.x * blockDim.x)
    d[i] = (_Float16)s[i];
}

// ---------------------------------------------------------------- WMMA frags
// A fragment (16x32 f16): lane = hi*16 + m holds row m, K in
// [k0+hi*8, +8) for VGPR0..3 and [k0+16+hi*8, +8) for VGPR4..7.
__device__ __forceinline__ v16h load_frag_a(const _Float16* __restrict__ row,
                                            int k0, int hi) {
  const _Float16* p = row + k0 + hi * 8;
  v8h lo = *(const v8h*)p;
  v8h up = *(const v8h*)(p + 16);
  v16h r;
#pragma unroll
  for (int i = 0; i < 8; ++i) { r[i] = lo[i]; r[i + 8] = up[i]; }
  return r;
}
// B fragment (32x16 f16): lane = hi*16 + n holds column n (= row n of W,
// since B = W^T), K in [k0+hi*8, +8) then [k0+16+hi*8, +8) (same as A).
__device__ __forceinline__ v16h load_frag_b(const _Float16* __restrict__ row,
                                            int k0, int hi) {
  const _Float16* p = row + k0 + hi * 8;
  v8h lo = *(const v8h*)p;
  v8h up = *(const v8h*)(p + 16);
  v16h r;
#pragma unroll
  for (int i = 0; i < 8; ++i) { r[i] = lo[i]; r[i + 8] = up[i]; }
  return r;
}

#if USE_TDM
// ---------------------------------------------------------------- TDM helper
// 2-D tile load: tile_rows rows of tile_k f16 elements, global row stride
// row_stride elements, into LDS at lds_off with 4-DWORD padding every
// 32 DWORDs (=> LDS row stride LDR halves, bank-conflict-free frag reads).
__device__ __forceinline__ uint32_t lds_off_of(const _Float16* p) {
  return (uint32_t)(uintptr_t)p;  // low 32 bits of flat shared addr = LDS offset
}

__device__ __forceinline__ void tdm_load_2d(const _Float16* gptr,
                                            uint32_t lds_off, int tile_k,
                                            int tile_rows, int row_stride) {
  uint32_t g0[4] = {0u, 0u, 0u, 0u};
  uint32_t g1[8] = {0u, 0u, 0u, 0u, 0u, 0u, 0u, 0u};
  uint64_t ga = (uint64_t)(uintptr_t)gptr;
  g0[0] = 1u;                                   // count=1 (valid user D#)
  g0[1] = lds_off;                              // lds_addr
  g0[2] = (uint32_t)ga;                         // global_addr[31:0]
  g0[3] = (uint32_t)((ga >> 32) & 0x1FFFFFFu)   // global_addr[56:32]
          | (2u << 30);                         // type = 2 ("image")
  // data_size=1 (2B), pad_enable, pad_interval code 4 (32 DWORDs = 64 elems),
  // pad_amount code 3 (4 DWORDs = 8 halves) -> LDS row stride = 72 halves.
  g1[0] = (1u << 16) | (1u << 20) | (4u << 22) | (3u << 25);
  uint64_t td0 = (uint64_t)tile_k;              // tensor_dim0, bits 79:48
  g1[1] |= (uint32_t)(td0 << 16);
  g1[2] |= (uint32_t)((td0 >> 16) & 0xFFFFu);
  uint64_t td1 = (uint64_t)tile_rows;           // tensor_dim1, bits 111:80
  g1[2] |= (uint32_t)(td1 << 16);
  g1[3] |= (uint32_t)((td1 >> 16) & 0xFFFFu);
  g1[3] |= ((uint32_t)tile_k & 0xFFFFu) << 16;  // tile_dim0, bits 127:112
  g1[4] |= (uint32_t)tile_rows & 0xFFFFu;       // tile_dim1, bits 143:128
  uint64_t st0 = (uint64_t)row_stride;          // tensor_dim0_stride 207:160
  g1[5] = (uint32_t)st0;
  g1[6] |= (uint32_t)((st0 >> 32) & 0xFFFFu);
  u32x4 G0 = {g0[0], g0[1], g0[2], g0[3]};
  i32x8_t G1 = {(int)g1[0], (int)g1[1], (int)g1[2], (int)g1[3],
                (int)g1[4], (int)g1[5], (int)g1[6], (int)g1[7]};
  i32x4_t Z4 = {0, 0, 0, 0};
  i32x8_t Z8 = {0, 0, 0, 0, 0, 0, 0, 0};
  __builtin_amdgcn_tensor_load_to_lds(G0, G1, Z4, Z4, Z8, 0);
}
#endif  // USE_TDM

// C = A[M,K] * W[N,K]^T + bias, epilogues:
//   EP=0 none, EP=1 sigmoid, EP=2 *scale, EP=3 +addend[r*N+c]
template <int EP>
__global__ __launch_bounds__(256) void gemm_f16_kernel(
    const _Float16* __restrict__ A, const _Float16* __restrict__ W,
    const float* __restrict__ bias, const float* __restrict__ addend,
    float* __restrict__ C, _Float16* __restrict__ Ch, int M, int N, int K,
    float scale) {
  __shared__ _Float16 As[2][128 * LDR];  // 36 KB
  __shared__ _Float16 Bs[2][64 * LDR];   // 18 KB

  const int tid  = threadIdx.x;
  const int lane = tid & 31, wave = tid >> 5;
  const int hi = lane >> 4, m = lane & 15;
  const int rowBlock = blockIdx.y * 128;
  const int colBase  = blockIdx.x * 64;
  const int S = K / KC;

  v8f acc0 = {}, acc1 = {}, acc2 = {}, acc3 = {};

#if USE_TDM
  if (wave == 0) {
    tdm_load_2d(A + (size_t)rowBlock * K, lds_off_of(&As[0][0]), KC, 128, K);
    tdm_load_2d(W + (size_t)colBase * K, lds_off_of(&Bs[0][0]), KC, 64, K);
  }
  for (int s = 0; s < S; ++s) {
    const int cur = s & 1;
    if (wave == 0) __builtin_amdgcn_s_wait_tensorcnt(0);  // stage s landed
    __syncthreads();
    if (wave == 0 && s + 1 < S) {  // DMA next stage while computing this one
      const int ks = (s + 1) * KC;
      tdm_load_2d(A + (size_t)rowBlock * K + ks, lds_off_of(&As[cur ^ 1][0]),
                  KC, 128, K);
      tdm_load_2d(W + (size_t)colBase * K + ks, lds_off_of(&Bs[cur ^ 1][0]),
                  KC, 64, K);
    }
    const _Float16* arow = &As[cur][(wave * 16 + m) * LDR];
    const _Float16* b0r = &Bs[cur][(0 + m) * LDR];
    const _Float16* b1r = &Bs[cur][(16 + m) * LDR];
    const _Float16* b2r = &Bs[cur][(32 + m) * LDR];
    const _Float16* b3r = &Bs[cur][(48 + m) * LDR];
#pragma unroll
    for (int kk = 0; kk < KC; kk += 32) {
      v16h a  = load_frag_a(arow, kk, hi);
      v16h b0 = load_frag_b(b0r, kk, hi);
      v16h b1 = load_frag_b(b1r, kk, hi);
      v16h b2 = load_frag_b(b2r, kk, hi);
      v16h b3 = load_frag_b(b3r, kk, hi);
      acc0 = __builtin_amdgcn_wmma_f32_16x16x32_f16(false, a, false, b0,
                                                    (short)0, acc0, false, false);
      acc1 = __builtin_amdgcn_wmma_f32_16x16x32_f16(false, a, false, b1,
                                                    (short)0, acc1, false, false);
      acc2 = __builtin_amdgcn_wmma_f32_16x16x32_f16(false, a, false, b2,
                                                    (short)0, acc2, false, false);
      acc3 = __builtin_amdgcn_wmma_f32_16x16x32_f16(false, a, false, b3,
                                                    (short)0, acc3, false, false);
    }
  }
#else
  // Fallback: cooperative global->LDS staging (same padded layout).
  for (int s = 0; s < S; ++s) {
    const int cur = s & 1;
    const int ks = s * KC;
    for (int c = tid; c < 128 * (KC / 8); c += 256) {
      int r = c >> 3, cc = (c & 7) * 8;
      *(v8h*)&As[cur][r * LDR + cc] =
          *(const v8h*)&A[(size_t)(rowBlock + r) * K + ks + cc];
    }
    for (int c = tid; c < 64 * (KC / 8); c += 256) {
      int r = c >> 3, cc = (c & 7) * 8;
      *(v8h*)&Bs[cur][r * LDR + cc] =
          *(const v8h*)&W[(size_t)(colBase + r) * K + ks + cc];
    }
    __syncthreads();
    const _Float16* arow = &As[cur][(wave * 16 + m) * LDR];
    const _Float16* b0r = &Bs[cur][(0 + m) * LDR];
    const _Float16* b1r = &Bs[cur][(16 + m) * LDR];
    const _Float16* b2r = &Bs[cur][(32 + m) * LDR];
    const _Float16* b3r = &Bs[cur][(48 + m) * LDR];
#pragma unroll
    for (int kk = 0; kk < KC; kk += 32) {
      v16h a  = load_frag_a(arow, kk, hi);
      v16h b0 = load_frag_b(b0r, kk, hi);
      v16h b1 = load_frag_b(b1r, kk, hi);
      v16h b2 = load_frag_b(b2r, kk, hi);
      v16h b3 = load_frag_b(b3r, kk, hi);
      acc0 = __builtin_amdgcn_wmma_f32_16x16x32_f16(false, a, false, b0,
                                                    (short)0, acc0, false, false);
      acc1 = __builtin_amdgcn_wmma_f32_16x16x32_f16(false, a, false, b1,
                                                    (short)0, acc1, false, false);
      acc2 = __builtin_amdgcn_wmma_f32_16x16x32_f16(false, a, false, b2,
                                                    (short)0, acc2, false, false);
      acc3 = __builtin_amdgcn_wmma_f32_16x16x32_f16(false, a, false, b3,
                                                    (short)0, acc3, false, false);
    }
    __syncthreads();
  }
#endif

  const float bia0 = bias ? bias[colBase + 0  + m] : 0.f;
  const float bia1 = bias ? bias[colBase + 16 + m] : 0.f;
  const float bia2 = bias ? bias[colBase + 32 + m] : 0.f;
  const float bia3 = bias ? bias[colBase + 48 + m] : 0.f;

#pragma unroll
  for (int v = 0; v < 8; ++v) {
    int r = rowBlock + wave * 16 + 8 * hi + v;  // C/D: M = vgpr + 8*hi
    size_t base = (size_t)r * N + colBase + m;
    float x0 = acc0[v] + bia0;
    float x1 = acc1[v] + bia1;
    float x2 = acc2[v] + bia2;
    float x3 = acc3[v] + bia3;
    if (EP == 1) {
      x0 = 1.f / (1.f + expf(-x0)); x1 = 1.f / (1.f + expf(-x1));
      x2 = 1.f / (1.f + expf(-x2)); x3 = 1.f / (1.f + expf(-x3));
    } else if (EP == 2) {
      x0 *= scale; x1 *= scale; x2 *= scale; x3 *= scale;
    } else if (EP == 3) {
      x0 += addend[base];      x1 += addend[base + 16];
      x2 += addend[base + 32]; x3 += addend[base + 48];
    }
    C[base] = x0; C[base + 16] = x1; C[base + 32] = x2; C[base + 48] = x3;
    if (Ch) {
      Ch[base] = (_Float16)x0;      Ch[base + 16] = (_Float16)x1;
      Ch[base + 32] = (_Float16)x2; Ch[base + 48] = (_Float16)x3;
    }
  }
}

// ---------------------------------------------------------- causal conv+SiLU
__global__ __launch_bounds__(256) void conv_kernel(
    const float* __restrict__ xt, const float* __restrict__ cw,
    const float* __restrict__ cb, float* __restrict__ xc,
    _Float16* __restrict__ xch) {
  const int row = blockIdx.x, tid = threadIdx.x;
  const float w0 = cw[0], w1 = cw[1], w2 = cw[2], w3 = cw[3], bb = cb[0];
  const float* x = xt + (size_t)row * P_;
  for (int j = tid; j < P_; j += 256) {
    float a = bb + w3 * x[j];
    if (j >= 1) a += w2 * x[j - 1];
    if (j >= 2) a += w1 * x[j - 2];
    if (j >= 3) a += w0 * x[j - 3];
    float y = a / (1.f + expf(-a));  // silu
    xc[(size_t)row * P_ + j]  = y;
    xch[(size_t)row * P_ + j] = (_Float16)y;
  }
}

// ----------------------------------------------------------- i/f gate dots
__global__ __launch_bounds__(256) void gate_kernel(
    const float* __restrict__ xc, const float* __restrict__ Wi,
    const float* __restrict__ bi, const float* __restrict__ Wf,
    const float* __restrict__ bf, const float* __restrict__ m_tm1,
    float* __restrict__ ig, float* __restrict__ fg, float* __restrict__ m_out) {
  __shared__ float red[256];
  const int bh = blockIdx.x, tid = threadIdx.x;
  const int b = bh >> 3, h = bh & 7;
  const float* x  = xc + (size_t)b * P_;
  const float* wi = Wi + (size_t)h * P_;
  const float* wf = Wf + (size_t)h * P_;
  float si = 0.f, sf = 0.f;
  for (int j = tid; j < P_; j += 256) {
    float xv = x[j];
    si += xv * wi[j];
    sf += xv * wf[j];
  }
  float ti = block_reduce_sum(si, red, tid);
  float tf = block_reduce_sum(sf, red, tid);
  if (tid == 0) {
    float it = ti + bi[h], ft = tf + bf[h];
    float mo = m_tm1[bh];
    float mt = fmaxf(ft + mo, it);
    m_out[bh] = mt;
    ig[bh] = expf(it - mt);
    fg[bh] = expf(ft - mt + mo);
  }
}

// ---------- fused cell: c update + num + n_t/den + gate + GroupNorm per (b,h)
__global__ __launch_bounds__(256) void cell_kernel(
    const float* __restrict__ c_tm1, const float* __restrict__ n_tm1,
    const float* __restrict__ qv, const float* __restrict__ kv,
    const float* __restrict__ vv, const float* __restrict__ ov,
    const float* __restrict__ ig, const float* __restrict__ fg,
    const float* __restrict__ gng, const float* __restrict__ gnb,
    float* __restrict__ c_out, float* __restrict__ n_out,
    float* __restrict__ hn) {
  __shared__ __align__(16) float q_s[HD_];
  __shared__ __align__(16) float k_s[HD_];
  __shared__ float v_s[HD_];
  __shared__ float num_s[HD_];
  __shared__ float red[256];

  const int bh = blockIdx.x, tid = threadIdx.x;
  const int b = bh >> 3, h = bh & 7;
  if (tid < HD_) {
    size_t gi = (size_t)b * HID_ + h * HD_ + tid;
    q_s[tid] = qv[gi];
    k_s[tid] = kv[gi];
    v_s[tid] = vv[gi];
  }
  __syncthreads();

  const float f = fg[bh], i = ig[bh];
  const float* crow = c_tm1 + (size_t)bh * HD_ * HD_;
  float* cout = c_out + (size_t)bh * HD_ * HD_;
  const int wave = tid >> 5, lane = tid & 31;
  const float4 k4 = ((const float4*)k_s)[lane];
  const float4 q4 = ((const float4*)q_s)[lane];

  for (int r = wave * 16; r < wave * 16 + 16; ++r) {
    float4 co = *(const float4*)(crow + (size_t)r * HD_ + lane * 4);
    if (r + 1 < HD_)
      __builtin_prefetch(crow + (size_t)(r + 1) * HD_ + lane * 4, 0, 0);
    float iv = i * v_s[r];
    float4 cn;
    cn.x = f * co.x + iv * k4.x;
    cn.y = f * co.y + iv * k4.y;
    cn.z = f * co.z + iv * k4.z;
    cn.w = f * co.w + iv * k4.w;
    *(float4*)(cout + (size_t)r * HD_ + lane * 4) = cn;
    float p = cn.x * q4.x + cn.y * q4.y + cn.z * q4.z + cn.w * q4.w;
#pragma unroll
    for (int off = 16; off > 0; off >>= 1) p += __shfl_down(p, off, 32);
    if (lane == 0) num_s[r] = p;
  }
  __syncthreads();

  float nd = 0.f;
  if (tid < HD_) {
    float nt = f * n_tm1[(size_t)bh * HD_ + tid] + i * k_s[tid];
    n_out[(size_t)bh * HD_ + tid] = nt;
    nd = nt * q_s[tid];
  }
  float den = fmaxf(block_reduce_sum(nd, red, tid), 1.f);

  float hv = 0.f;
  if (tid < HD_)
    hv = ov[(size_t)b * HID_ + h * HD_ + tid] * (num_s[tid] / den);
  float sum = block_reduce_sum(hv, red, tid);
  float sq  = block_reduce_sum(hv * hv, red, tid);
  float mu  = sum * (1.f / HD_);
  float var = sq * (1.f / HD_) - mu * mu;
  float inv = rsqrtf(var + EPS_);
  if (tid < HD_) {
    int g = h * HD_ + tid;
    hn[(size_t)b * HID_ + g] = (hv - mu) * inv * gng[g] + gnb[g];
  }
}

// --------------------------------------------- (hn + skip) * silu(r_t) -> f16
__global__ void fuse_kernel(const float* __restrict__ hn,
                            const float* __restrict__ skip,
                            const float* __restrict__ rt,
                            _Float16* __restrict__ outh, int n) {
  for (int idx = blockIdx.x * blockDim.x + threadIdx.x; idx < n;
       idx += gridDim.x * blockDim.x) {
    float r = rt[idx];
    float sil = r / (1.f + expf(-r));
    outh[idx] = (_Float16)((hn[idx] + skip[idx]) * sil);
  }
}

// ================================================================= launcher
extern "C" void kernel_launch(void* const* d_in, const int* in_sizes, int n_in,
                              void* d_out, int out_size, void* d_ws,
                              size_t ws_size, hipStream_t stream) {
  (void)in_sizes; (void)n_in; (void)out_size; (void)ws_size;
  const float* seq    = (const float*)d_in[0];
  const float* c_tm1  = (const float*)d_in[1];
  const float* n_tm1  = (const float*)d_in[2];
  const float* m_tm1  = (const float*)d_in[3];
  const float* ln_g   = (const float*)d_in[4];
  const float* ln_b   = (const float*)d_in[5];
  const float* gn_g   = (const float*)d_in[6];
  const float* gn_b   = (const float*)d_in[7];
  const float* W_upl  = (const float*)d_in[8];
  const float* b_upl  = (const float*)d_in[9];
  const float* W_upr  = (const float*)d_in[10];
  const float* b_upr  = (const float*)d_in[11];
  const float* W_down = (const float*)d_in[12];
  const float* b_down = (const float*)d_in[13];
  const float* W_last = (const float*)d_in[14];
  const float* b_last = (const float*)d_in[15];
  const float* conv_w = (const float*)d_in[16];
  const float* conv_b = (const float*)d_in[17];
  const float* W_skip = (const float*)d_in[18];
  const float* W_i    = (const float*)d_in[19];
  const float* b_i    = (const float*)d_in[20];
  const float* W_f    = (const float*)d_in[21];
  const float* b_f    = (const float*)d_in[22];
  const float* W_o    = (const float*)d_in[23];
  const float* b_o    = (const float*)d_in[24];
  const float* W_q    = (const float*)d_in[25];
  const float* b_q    = (const float*)d_in[26];
  const float* W_k    = (const float*)d_in[27];
  const float* b_k    = (const float*)d_in[28];
  const float* W_v    = (const float*)d_in[29];
  const float* b_v    = (const float*)d_in[30];

  float* out_final = (float*)d_out;                        // [B,O]
  float* out_c = out_final + (size_t)B_ * O_;              // [B,H,HD,HD]
  float* out_n = out_c + (size_t)B_ * H_ * HD_ * HD_;      // [B,H,HD]
  float* out_m = out_n + (size_t)B_ * H_ * HD_;            // [B,H]

  char* ws = (char*)d_ws;
  size_t off = 0;
  auto alloc = [&](size_t bytes) -> char* {
    size_t a = (off + 255) & ~(size_t)255;
    off = a + bytes;
    return ws + a;
  };
  float*    xn_f   = (float*)alloc((size_t)B_ * D_ * 4);
  _Float16* xn_h   = (_Float16*)alloc((size_t)B_ * D_ * 2);
  float*    xt_f   = (float*)alloc((size_t)B_ * P_ * 4);
  _Float16* xt_h   = (_Float16*)alloc((size_t)B_ * P_ * 2);
  float*    rt_f   = (float*)alloc((size_t)B_ * HID_ * 4);
  float*    xc_f   = (float*)alloc((size_t)B_ * P_ * 4);
  _Float16* xc_h   = (_Float16*)alloc((size_t)B_ * P_ * 2);
  float*    q_f    = (float*)alloc((size_t)B_ * HID_ * 4);
  float*    k_f    = (float*)alloc((size_t)B_ * HID_ * 4);
  float*    v_f    = (float*)alloc((size_t)B_ * HID_ * 4);
  float*    o_f    = (float*)alloc((size_t)B_ * HID_ * 4);
  float*    skip_f = (float*)alloc((size_t)B_ * HID_ * 4);
  float*    hn_f   = (float*)alloc((size_t)B_ * HID_ * 4);
  _Float16* fuse_h = (_Float16*)alloc((size_t)B_ * HID_ * 2);
  float*    y2_f   = (float*)alloc((size_t)B_ * D_ * 4);
  _Float16* y2_h   = (_Float16*)alloc((size_t)B_ * D_ * 2);
  float*    ig_f   = (float*)alloc((size_t)B_ * H_ * 4);
  float*    fg_f   = (float*)alloc((size_t)B_ * H_ * 4);
  _Float16* wupl_h  = (_Float16*)alloc((size_t)P_ * D_ * 2);
  _Float16* wupr_h  = (_Float16*)alloc((size_t)HID_ * D_ * 2);
  _Float16* wq_h    = (_Float16*)alloc((size_t)HID_ * P_ * 2);
  _Float16* wk_h    = (_Float16*)alloc((size_t)HID_ * P_ * 2);
  _Float16* wv_h    = (_Float16*)alloc((size_t)HID_ * P_ * 2);
  _Float16* wo_h    = (_Float16*)alloc((size_t)HID_ * P_ * 2);
  _Float16* wskip_h = (_Float16*)alloc((size_t)HID_ * P_ * 2);
  _Float16* wdown_h = (_Float16*)alloc((size_t)D_ * HID_ * 2);
  _Float16* wlast_h = (_Float16*)alloc((size_t)O_ * D_ * 2);

  auto cvt = [&](const float* s, _Float16* d, size_t n) {
    int blocks = (int)((n + 255) / 256);
    if (blocks > 4096) blocks = 4096;
    cvt_kernel<<<blocks, 256, 0, stream>>>(s, d, (int)n);
  };

  // Stage 0: LN + weight conversions
  ln_kernel<<<B_, 256, 0, stream>>>(seq, ln_g, ln_b, xn_f, xn_h);
  cvt(W_upl, wupl_h, (size_t)P_ * D_);
  cvt(W_upr, wupr_h, (size_t)HID_ * D_);
  cvt(W_q, wq_h, (size_t)HID_ * P_);
  cvt(W_k, wk_h, (size_t)HID_ * P_);
  cvt(W_v, wv_h, (size_t)HID_ * P_);
  cvt(W_o, wo_h, (size_t)HID_ * P_);
  cvt(W_skip, wskip_h, (size_t)HID_ * P_);
  cvt(W_down, wdown_h, (size_t)D_ * HID_);
  cvt(W_last, wlast_h, (size_t)O_ * D_);

  const dim3 blk(256);
  const dim3 gP(P_ / 64, B_ / 128);    // N=2048
  const dim3 gH(HID_ / 64, B_ / 128);  // N=1024

  // Stage 1: up projections
  gemm_f16_kernel<0><<<gP, blk, 0, stream>>>(xn_h, wupl_h, b_upl, nullptr,
                                             xt_f, xt_h, B_, P_, D_, 1.f);
  gemm_f16_kernel<0><<<gH, blk, 0, stream>>>(xn_h, wupr_h, b_upr, nullptr,
                                             rt_f, nullptr, B_, HID_, D_, 1.f);
  // Stage 2: causal conv + silu
  conv_kernel<<<B_, blk, 0, stream>>>(xt_f, conv_w, conv_b, xc_f, xc_h);

  // Stage 3: q/k/v/o/skip projections
  gemm_f16_kernel<0><<<gH, blk, 0, stream>>>(xc_h, wq_h, b_q, nullptr, q_f,
                                             nullptr, B_, HID_, P_, 1.f);
  gemm_f16_kernel<2><<<gH, blk, 0, stream>>>(xc_h, wk_h, b_k, nullptr, k_f,
                                             nullptr, B_, HID_, P_,
                                             0.08838834764831845f);  // 1/sqrt(128)
  gemm_f16_kernel<0><<<gH, blk, 0, stream>>>(xt_h, wv_h, b_v, nullptr, v_f,
                                             nullptr, B_, HID_, P_, 1.f);
  gemm_f16_kernel<1><<<gH, blk, 0, stream>>>(xt_h, wo_h, b_o, nullptr, o_f,
                                             nullptr, B_, HID_, P_, 1.f);
  gemm_f16_kernel<0><<<gH, blk, 0, stream>>>(xc_h, wskip_h, nullptr, nullptr,
                                             skip_f, nullptr, B_, HID_, P_, 1.f);

  // Stage 4: gates, fused matrix-memory cell (c_t/n_t/m_t + GroupNorm)
  gate_kernel<<<B_ * H_, blk, 0, stream>>>(xc_f, W_i, b_i, W_f, b_f, m_tm1,
                                           ig_f, fg_f, out_m);
  cell_kernel<<<B_ * H_, blk, 0, stream>>>(c_tm1, n_tm1, q_f, k_f, v_f, o_f,
                                           ig_f, fg_f, gn_g, gn_b, out_c,
                                           out_n, hn_f);

  // Stage 5: skip merge, down projection (+x_n residual), last projection
  fuse_kernel<<<1024, blk, 0, stream>>>(hn_f, skip_f, rt_f, fuse_h, B_ * HID_);
  gemm_f16_kernel<3><<<dim3(D_ / 64, B_ / 128), blk, 0, stream>>>(
      fuse_h, wdown_h, b_down, xn_f, y2_f, y2_h, B_, D_, HID_, 1.f);
  gemm_f16_kernel<0><<<dim3(O_ / 64, B_ / 128), blk, 0, stream>>>(
      y2_h, wlast_h, b_last, nullptr, out_final, nullptr, B_, O_, D_, 1.f);
}
